// Model_24498493456441
// MI455X (gfx1250) — matrix operand order repaired and checked by reading, not compiled, not run on hardware
//
#include <hip/hip_runtime.h>
#include <hip/hip_bf16.h>

// ---------------------------------------------------------------------------
// MS-G3D forward for MI455X (gfx1250, wave32). All channel-contraction GEMMs
// run through v_wmma_f32_16x16x32_f16 with fused bias/BN/ReLU/residual
// epilogues; weights are pre-converted to f16 once per layer; graph
// aggregation / attention / pooling glue is VALU.
// ---------------------------------------------------------------------------

typedef __attribute__((ext_vector_type(16))) _Float16     v16h;
typedef __attribute__((ext_vector_type(8)))  float        v8f;
typedef __attribute__((ext_vector_type(4)))  unsigned int v4u;

#define VJ 17   // joints

// B fragment loader: 16 consecutive halves of one weight row.
// AL=true: two 16B vector loads (row 16B-aligned when K%8==0).
template <bool AL>
__device__ __forceinline__ v16h load_bfrag(const _Float16* __restrict__ wr)
{
  if constexpr (AL) {
    union { v16h h; v4u u[2]; } t;
    t.u[0] = *(const v4u*)wr;
    t.u[1] = *(const v4u*)(wr + 8);
    return t.h;
  } else {
    v16h b;
#pragma unroll
    for (int e = 0; e < 16; ++e) b[e] = wr[e];
    return b;
  }
}

// Main K loop body: load A fragment + all 4 B fragments, then issue the 4
// WMMAs back-to-back (only the last pays the WMMA->VALU WAR hazard).
template <bool AL>
__device__ __forceinline__ void gemm_mainloop(const float* __restrict__ Xa, int astride,
                                              const _Float16* const* Wr, int Kmain,
                                              int half, v8f* acc)
{
  for (int k0 = 0; k0 < Kmain; k0 += 32) {
    const int ka0 = k0 + (half << 4);
    v16h a;
#pragma unroll
    for (int e = 0; e < 8; ++e) {
      a[e]     = (_Float16)Xa[(size_t)(ka0 + e) * astride];
      a[e + 8] = (_Float16)Xa[(size_t)(ka0 + 8 + e) * astride];
    }
    v16h bf[4];
#pragma unroll
    for (int j = 0; j < 4; ++j)
      bf[j] = load_bfrag<AL>(Wr[j] + k0 + (half << 4));
#pragma unroll
    for (int j = 0; j < 4; ++j)
      acc[j] = __builtin_amdgcn_wmma_f32_16x16x32_f16(
          false, a, false, bf[j], (short)0, acc[j], false, false);
  }
}

// ---------------------------------------------------------------------------
// Generic WMMA GEMM:  Y = epilogue( X[M x K](f32) * Wh[N x K](f16)^T )
// xl==0: X token-major; xl==1: X NCTV (xC channels, xTV=T*V)
// yl==0: Y token-major (yCt cols, offset yco); yl==1: Y NCTV
// flags: 1=bias 2=bn-affine 4=relu 8=residual (res indexed like Y)
// One wave computes a 16(M) x 64(N) strip. Main loop is branch-free: OOB
// rows/cols load from clamped addresses (their accumulators are never
// stored); the K remainder is one masked tail step.
// ---------------------------------------------------------------------------
__global__ __launch_bounds__(128)
void k_gemm(const float* __restrict__ X, const _Float16* __restrict__ Wh,
            const float* __restrict__ bias, const float* __restrict__ bng,
            const float* __restrict__ bnb, const float* __restrict__ res,
            float* __restrict__ Y,
            int Mrows, int N, int K,
            int xl, int xC, int xTV,
            int yl, int yCt, int yco, int yTV,
            int flags)
{
  const int lane = threadIdx.x & 31;
  const int wave = threadIdx.x >> 5;
  const int mtiles  = (Mrows + 15) >> 4;
  const int nstrips = (N + 63) >> 6;
  const int wg = blockIdx.x * 4 + wave;
  if (wg >= mtiles * nstrips) return;
  const int mt = wg % mtiles;
  const int ns = wg / mtiles;
  const int m0 = mt << 4, n0 = ns << 6;
  const int half = lane >> 4, l15 = lane & 15;

  // A-side base pointer (row clamped: dead rows never stored)
  const int arow = min(m0 + l15, Mrows - 1);
  int astride;
  const float* Xa;
  if (xl == 0) { Xa = X + (size_t)arow * K; astride = 1; }
  else {
    int nn = arow / xTV, rr = arow - nn * xTV;
    Xa = X + (size_t)nn * xC * xTV + rr; astride = xTV;
  }
  // B-side row pointers (col clamped: dead cols never stored)
  const _Float16* Wr[4];
#pragma unroll
  for (int j = 0; j < 4; ++j) {
    int bcol = min(n0 + (j << 4) + l15, N - 1);
    Wr[j] = Wh + (size_t)bcol * K;
  }

  v8f acc[4] = {};
  const int Kmain = K & ~31;

  if ((K & 7) == 0) gemm_mainloop<true >(Xa, astride, Wr, Kmain, half, acc);
  else              gemm_mainloop<false>(Xa, astride, Wr, Kmain, half, acc);

  // ---- masked tail (clamped addresses + selects, no branches) ----
  if (Kmain < K) {
    const int ka0 = Kmain + (half << 4);
    v16h a;
#pragma unroll
    for (int e = 0; e < 8; ++e) {
      int k1 = ka0 + e, k2 = ka0 + 8 + e;
      float v1 = Xa[(size_t)min(k1, K - 1) * astride];
      float v2 = Xa[(size_t)min(k2, K - 1) * astride];
      a[e]     = (_Float16)(k1 < K ? v1 : 0.f);
      a[e + 8] = (_Float16)(k2 < K ? v2 : 0.f);
    }
    const int kb0 = Kmain + (half << 4);
    v16h bf[4];
#pragma unroll
    for (int j = 0; j < 4; ++j) {
#pragma unroll
      for (int e = 0; e < 16; ++e) {
        int kk = kb0 + e;
        _Float16 v = Wr[j][min(kk, K - 1)];
        bf[j][e] = (kk < K) ? v : (_Float16)0.f;
      }
    }
#pragma unroll
    for (int j = 0; j < 4; ++j)
      acc[j] = __builtin_amdgcn_wmma_f32_16x16x32_f16(
          false, a, false, bf[j], (short)0, acc[j], false, false);
  }

  // ---- epilogue. C/D layout: VGPR r -> row m0 + r + 8*half, col = n0+16j+l15
#pragma unroll
  for (int j = 0; j < 4; ++j) {
    const int col = n0 + (j << 4) + l15;
    if (col >= N) continue;
    const float bi = (flags & 1) ? bias[col] : 0.f;
    const float g  = (flags & 2) ? bng[col]  : 1.f;
    const float bb = (flags & 2) ? bnb[col]  : 0.f;
#pragma unroll
    for (int r = 0; r < 8; ++r) {
      const int row = m0 + r + (half << 3);
      if (row >= Mrows) continue;
      float y = acc[j][r] + bi;
      y = y * g + bb;
      size_t idx;
      if (yl == 0) idx = (size_t)row * yCt + yco + col;
      else {
        int nn = row / yTV, rr = row - nn * yTV;
        idx = ((size_t)nn * yCt + yco + col) * yTV + rr;
      }
      if (flags & 8) y += res[idx];
      if (flags & 4) y = fmaxf(y, 0.f);
      Y[idx] = y;
    }
  }
}

// f32 -> f16 weight staging
__global__ void k_f16cvt(const float* __restrict__ s, _Float16* __restrict__ d, int n)
{
  int i = blockIdx.x * blockDim.x + threadIdx.x;
  if (i < n) d[i] = (_Float16)s[i];
}

// ---------------------------------------------------------------------------
// Build multiscale normalized adjacency on-device (deterministic each call).
// out[k] = norm( reach_k - reach_{k-1} + I ), reach via boolean powers in LDS.
// tile>1 => base = tile(Abin + I, (tile,tile)). Single block.
// ---------------------------------------------------------------------------
__global__ void k_build_adj(float* __restrict__ out, int tile, int S)
{
  __shared__ float Rp[85 * 85];
  __shared__ float Rc[85 * 85];
  __shared__ float dsum[85];
  __shared__ int   adj[17 * 17];
  const int W = VJ * tile;
  const int WW = W * W;
  const int tid = threadIdx.x;
  for (int i = tid; i < 289; i += blockDim.x) adj[i] = 0;
  __syncthreads();
  if (tid == 0) {
    const int E[18][2] = {{0,1},{0,2},{1,3},{2,4},{5,7},{7,9},{6,8},{8,10},{5,6},
                          {5,11},{6,12},{11,12},{11,13},{13,15},{12,14},{14,16},{0,5},{0,6}};
    for (int e = 0; e < 18; ++e) {
      adj[E[e][0] * 17 + E[e][1]] = 1;
      adj[E[e][1] * 17 + E[e][0]] = 1;
    }
  }
  __syncthreads();
  for (int idx = tid; idx < WW; idx += blockDim.x) {
    int i = idx / W, j = idx % W;
    float I = (i == j) ? 1.f : 0.f;
    Rp[idx] = I;
    out[idx] = I;            // scale 0: norm(I) = I
  }
  __syncthreads();
  for (int k = 1; k < S; ++k) {
    for (int idx = tid; idx < WW; idx += blockDim.x) {
      int i = idx / W, j = idx % W;
      float r = Rp[idx];
      if (r == 0.f) {
        int jb = j % VJ;
        for (int u = 0; u < W; ++u) {
          if (Rp[i * W + u] > 0.f) {
            int ub = u % VJ;
            int base = adj[ub * 17 + jb] | ((tile > 1 && ub == jb) ? 1 : 0);
            if (u == j || base) { r = 1.f; break; }
          }
        }
      }
      Rc[idx] = r;
    }
    __syncthreads();
    for (int i = tid; i < W; i += blockDim.x) {
      float s = 0.f;
      for (int j = 0; j < W; ++j)
        s += Rc[i * W + j] - Rp[i * W + j] + ((i == j) ? 1.f : 0.f);
      dsum[i] = s;
    }
    __syncthreads();
    for (int idx = tid; idx < WW; idx += blockDim.x) {
      int i = idx / W, j = idx % W;
      float ak = Rc[idx] - Rp[idx] + ((i == j) ? 1.f : 0.f);
      float di = dsum[i] > 0.f ? rsqrtf(dsum[i]) : 0.f;
      float dj = dsum[j] > 0.f ? rsqrtf(dsum[j]) : 0.f;
      out[(size_t)k * WW + idx] = di * ak * dj;
    }
    __syncthreads();
    for (int idx = tid; idx < WW; idx += blockDim.x) Rp[idx] = Rc[idx];
    __syncthreads();
  }
}

// ---------------------------------------------------------------------------
// Small glue kernels
// ---------------------------------------------------------------------------
__global__ void k_data_bn(const float* __restrict__ x, const float* __restrict__ g,
                          const float* __restrict__ b, float* __restrict__ y)
{
  int i = blockIdx.x * blockDim.x + threadIdx.x;
  const int total = 16 * 3 * 256 * VJ;
  if (i >= total) return;
  int v = i % VJ; int t = (i / VJ) % 256; int c = (i / (VJ * 256)) % 3; int nm = i / (VJ * 256 * 3);
  int n = nm >> 1, m = nm & 1;
  float xv = x[((((size_t)n * 3 + c) * 256 + t) * VJ + v) * 2 + m];
  int ai = (m * VJ + v) * 3 + c;
  y[i] = xv * g[ai] + b[ai];
}

__global__ void k_axpy(float* __restrict__ o, const float* __restrict__ a,
                       const float* __restrict__ b, int n)
{
  int i = blockIdx.x * blockDim.x + threadIdx.x;
  if (i < n) o[i] = a[i] + b[i];
}

__global__ void k_add_inplace(float* __restrict__ a, const float* __restrict__ b, int n)
{
  int i = blockIdx.x * blockDim.x + threadIdx.x;
  if (i < n) a[i] += b[i];
}

__global__ void k_add_relu(const float* __restrict__ a, const float* __restrict__ b,
                           float* __restrict__ o, int n)
{
  int i = blockIdx.x * blockDim.x + threadIdx.x;
  if (i < n) o[i] = fmaxf(a[i] + b[i], 0.f);
}

// agg for one batch element: Y[s*C+c, t, v] = sum_u A[s*Vd+v, u] * X[c, t, u]
__global__ void k_adj_agg(const float* __restrict__ A, const float* __restrict__ X,
                          float* __restrict__ Y, int C, int T, int Vd, int S)
{
  int i = blockIdx.x * blockDim.x + threadIdx.x;
  int total = S * C * T * Vd;
  if (i >= total) return;
  int v = i % Vd; int t = (i / Vd) % T; int c = (i / (Vd * T)) % C; int s = i / (Vd * T * C);
  const float* Ar = A + (size_t)(s * Vd + v) * Vd;
  const float* Xr = X + (size_t)(c * T + t) * Vd;
  float acc = 0.f;
  for (int u = 0; u < Vd; ++u) acc += Ar[u] * Xr[u];
  Y[((size_t)(s * C + c) * T + t) * Vd + v] = acc;
}

// unfold one batch element: (C,Tin,V) -> (C,tout,wsz*V)
__global__ void k_unfold(const float* __restrict__ X, float* __restrict__ U,
                         int C, int Tin, int tout, int wsz, int stride, int pad)
{
  int i = blockIdx.x * blockDim.x + threadIdx.x;
  int total = C * tout * wsz * VJ;
  if (i >= total) return;
  int v = i % VJ; int w = (i / VJ) % wsz; int to = (i / (VJ * wsz)) % tout; int c = i / (VJ * wsz * tout);
  int t = to * stride + w - pad;
  float val = (t >= 0 && t < Tin) ? X[((size_t)c * Tin + t) * VJ + v] : 0.f;
  U[(((size_t)c * tout + to) * wsz + w) * VJ + v] = val;
}

// g3d output conv for one batch element: (eo,tout,wsz*V) -> (Co,tout,V), +bn2, acc
__global__ void k_g3d_out(const float* __restrict__ H, const float* __restrict__ ow,
                          const float* __restrict__ ob, const float* __restrict__ g,
                          const float* __restrict__ b, float* __restrict__ Y,
                          int eo, int Co, int tout, int wsz, int acc)
{
  int i = blockIdx.x * blockDim.x + threadIdx.x;
  int total = Co * tout * VJ;
  if (i >= total) return;
  int v = i % VJ; int t = (i / VJ) % tout; int o = i / (VJ * tout);
  float val = ob[o];
  for (int e = 0; e < eo; ++e) {
    const float* hr = H + ((size_t)e * tout + t) * (wsz * VJ) + v;
    const float* wr = ow + (size_t)(o * eo + e) * wsz;
    for (int w = 0; w < wsz; ++w) val += wr[w] * hr[w * VJ];
  }
  val = val * g[o] + b[o];
  size_t idx = ((size_t)o * tout + t) * VJ + v;
  if (acc) Y[idx] += val; else Y[idx] = val;
}

// dilated 3-tap temporal conv over full tensor (16,bc,T,V) -> slice of (16,Ctot,T,V)
__global__ void k_tconv(const float* __restrict__ X, const float* __restrict__ w,
                        const float* __restrict__ bt, const float* __restrict__ g,
                        const float* __restrict__ b, float* __restrict__ Y,
                        int bc, int T, int d, int Ctot, int coff)
{
  int i = blockIdx.x * blockDim.x + threadIdx.x;
  int total = 16 * bc * T * VJ;
  if (i >= total) return;
  int v = i % VJ; int t = (i / VJ) % T; int o = (i / (VJ * T)) % bc; int n = i / (VJ * T * bc);
  float val = bt[o];
  for (int ii = 0; ii < bc; ++ii) {
    const float* xr = X + ((size_t)(n * bc + ii) * T) * VJ + v;
    const float* wr = w + (size_t)(o * bc + ii) * 3;
    for (int j = 0; j < 3; ++j) {
      int t2 = t + (j - 1) * d;
      if (t2 >= 0 && t2 < T) val += wr[j] * xr[(size_t)t2 * VJ];
    }
  }
  val = val * g[o] + b[o];
  Y[((size_t)(n * Ctot + coff + o) * T + t) * VJ + v] = val;
}

// temporal maxpool-3 + bn2 into slice
__global__ void k_maxpool_bn(const float* __restrict__ X, const float* __restrict__ g,
                             const float* __restrict__ b, float* __restrict__ Y,
                             int bc, int T, int Ctot, int coff)
{
  int i = blockIdx.x * blockDim.x + threadIdx.x;
  int total = 16 * bc * T * VJ;
  if (i >= total) return;
  int v = i % VJ; int t = (i / VJ) % T; int o = (i / (VJ * T)) % bc; int n = i / (VJ * T * bc);
  const float* xr = X + ((size_t)(n * bc + o) * T) * VJ + v;
  float m = xr[(size_t)t * VJ];
  if (t > 0)     m = fmaxf(m, xr[(size_t)(t - 1) * VJ]);
  if (t < T - 1) m = fmaxf(m, xr[(size_t)(t + 1) * VJ]);
  Y[((size_t)(n * Ctot + coff + o) * T + t) * VJ + v] = m * g[o] + b[o];
}

// (C,T,V) for one n -> token-major (V*T, C)
__global__ void k_nctv_to_tok(const float* __restrict__ X, float* __restrict__ tok,
                              int C, int T)
{
  int i = blockIdx.x * blockDim.x + threadIdx.x;
  int total = VJ * T * C;
  if (i >= total) return;
  int c = i % C; int t = (i / C) % T; int v = i / (C * T);
  tok[((size_t)v * T + t) * C + c] = X[((size_t)c * T + t) * VJ + v];
}

// LayerNorm per token row; one wave per row
__global__ void k_ln(const float* __restrict__ tok, const float* __restrict__ g,
                     const float* __restrict__ b, float* __restrict__ tn, int C)
{
  int row = blockIdx.x;
  int lane = threadIdx.x;
  const float* r = tok + (size_t)row * C;
  float s = 0.f, s2 = 0.f;
  for (int j = lane; j < C; j += 32) { float v = r[j]; s += v; s2 += v * v; }
  for (int off = 16; off > 0; off >>= 1) {
    s  += __shfl_xor(s,  off, 32);
    s2 += __shfl_xor(s2, off, 32);
  }
  float mean = s / C;
  float var  = s2 / C - mean * mean;
  float inv  = rsqrtf(var + 1e-5f);
  for (int j = lane; j < C; j += 32)
    tn[(size_t)row * C + j] = (r[j] - mean) * inv * g[j] + b[j];
}

// windowed multi-head attention; one wave per (window, head)
template <int WW, int DH>
__global__ void k_attn(const float* __restrict__ qkv, float* __restrict__ o, int c)
{
  int wid = blockIdx.x / 6, h = blockIdx.x % 6;
  int lane = threadIdx.x;
  if (lane >= WW) return;
  int c3 = 3 * c;
  int row0 = wid * WW;
  const float* qrow = qkv + (size_t)(row0 + lane) * c3 + h * DH;
  float q[DH];
#pragma unroll
  for (int d = 0; d < DH; ++d) q[d] = qrow[d];
  float s[WW];
  float mx = -1e30f;
  const float scl = rsqrtf((float)DH);
#pragma unroll
  for (int kk = 0; kk < WW; ++kk) {
    const float* krow = qkv + (size_t)(row0 + kk) * c3 + c + h * DH;
    float acc = 0.f;
#pragma unroll
    for (int d = 0; d < DH; ++d) acc += q[d] * krow[d];
    acc *= scl;
    s[kk] = acc; mx = fmaxf(mx, acc);
  }
  float sum = 0.f;
#pragma unroll
  for (int kk = 0; kk < WW; ++kk) { s[kk] = __expf(s[kk] - mx); sum += s[kk]; }
  float inv = 1.f / sum;
#pragma unroll
  for (int d = 0; d < DH; ++d) {
    float acc = 0.f;
#pragma unroll
    for (int kk = 0; kk < WW; ++kk)
      acc += s[kk] * qkv[(size_t)(row0 + kk) * c3 + 2 * c + h * DH + d];
    o[(size_t)(row0 + lane) * c + h * DH + d] = acc * inv;
  }
}

__global__ void k_tok_accum(const float* __restrict__ o2, const float* __restrict__ gate,
                            int gi, float* __restrict__ mix, int n, int init)
{
  int i = blockIdx.x * blockDim.x + threadIdx.x;
  if (i >= n) return;
  float s = gate[gi];
  mix[i] = init ? s * o2[i] : mix[i] + s * o2[i];
}

__global__ void k_mean_tv(const float* __restrict__ X, float* __restrict__ o, int C, int TV)
{
  int i = blockIdx.x * blockDim.x + threadIdx.x;
  if (i >= 16 * C) return;
  const float* xr = X + (size_t)i * TV;
  float s = 0.f;
  for (int r = 0; r < TV; ++r) s += xr[r];
  o[i] = s / TV;
}

__global__ void k_gate(const float* __restrict__ mean, const float* __restrict__ gw,
                       const float* __restrict__ gb, float* __restrict__ gate, int C)
{
  int n = threadIdx.x;
  if (n >= 16) return;
  float l0 = gb[0], l1 = gb[1];
  for (int c = 0; c < C; ++c) {
    float m = mean[(size_t)n * C + c];
    l0 += m * gw[c]; l1 += m * gw[C + c];
  }
  float mx = fmaxf(l0, l1);
  float e0 = __expf(l0 - mx), e1 = __expf(l1 - mx);
  float inv = 1.f / (e0 + e1);
  gate[n * 2 + 0] = e0 * inv;
  gate[n * 2 + 1] = e1 * inv;
}

// MoE post (one n): token-major y (V*T, Co) -> NCTV out (Co, Tout, V) with
// optional t-pair merge, bn, residual (NCTV), relu
__global__ void k_moe_post(const float* __restrict__ y, const float* __restrict__ g,
                           const float* __restrict__ b, const float* __restrict__ res,
                           float* __restrict__ o, int Co, int T, int merge, int relu,
                           int has_res)
{
  int Tout = merge ? T / 2 : T;
  int i = blockIdx.x * blockDim.x + threadIdx.x;
  int total = Co * Tout * VJ;
  if (i >= total) return;
  int v = i % VJ; int t = (i / VJ) % Tout; int c = i / (VJ * Tout);
  float val;
  if (merge) {
    float y0 = y[((size_t)v * T + 2 * t) * Co + c];
    float y1 = y[((size_t)v * T + 2 * t + 1) * Co + c];
    val = 0.5f * (y0 + y1);
  } else {
    val = y[((size_t)v * T + t) * Co + c];
  }
  val = val * g[c] + b[c];
  if (has_res) val += res[((size_t)c * T + t) * VJ + v];
  if (relu) val = fmaxf(val, 0.f);
  o[((size_t)c * Tout + t) * VJ + v] = val;
}

__global__ void k_pool_feat(const float* __restrict__ H, float* __restrict__ feat,
                            int C, int T)
{
  int i = blockIdx.x * blockDim.x + threadIdx.x;
  if (i >= 8 * C) return;
  int n = i / C, c = i % C;
  float s = 0.f;
  for (int m = 0; m < 2; ++m) {
    const float* hr = H + (((size_t)(n * 2 + m) * C + c) * T) * VJ;
    for (int r = 0; r < T * VJ; ++r) s += hr[r];
  }
  feat[i] = s / (2.f * T * VJ);
}

__global__ void k_fc(const float* __restrict__ feat, const float* __restrict__ w,
                     const float* __restrict__ b, float* __restrict__ out, int C)
{
  int i = blockIdx.x * blockDim.x + threadIdx.x;
  if (i >= 8 * 60) return;
  int n = i / 60, k = i % 60;
  float s = b[k];
  const float* f = feat + (size_t)n * C;
  const float* wr = w + (size_t)k * C;
  for (int c = 0; c < C; ++c) s += f[c] * wr[c];
  out[i] = s;
}

// ---------------------------------------------------------------------------
// Host side
// ---------------------------------------------------------------------------
struct BN  { const float *g, *b; };
struct GCN { const float *w, *b; BN bn; };
struct TBR { const float *w1, *b1; BN bn1; const float *wt, *bt; BN bn2; };
struct TCN { TBR br[4]; const float *mw1, *mb1; BN mbn1, mbn2; const float *ow, *ob; BN obn; };
struct G3D { const float *Ares, *w, *b; BN bn1; const float *ow, *ob; BN bn2; };
struct EXP { const float *lng, *lnb, *wqkv, *bqkv, *wo, *bo; };
struct MOE { EXP ex[2]; const float *gw, *gb, *pw, *pb; BN bn; };

static inline int cdiv(int a, int b) { return (a + b - 1) / b; }

static inline void cvt_w(hipStream_t st, const float* src, _Float16* dst, int n)
{
  k_f16cvt<<<cdiv(n, 256), 256, 0, st>>>(src, dst, n);
}

static inline void launch_gemm(hipStream_t st, const float* X, const _Float16* W,
                               const float* bias, const float* bg, const float* bb,
                               const float* res, float* Y,
                               int M, int N, int K,
                               int xl, int xC, int xTV,
                               int yl, int yCt, int yco, int yTV, int flags)
{
  int tiles  = ((M + 15) / 16) * ((N + 63) / 64);
  int blocks = (tiles + 3) / 4;
  k_gemm<<<blocks, 128, 0, st>>>(X, W, bias, bg, bb, res, Y, M, N, K,
                                 xl, xC, xTV, yl, yCt, yco, yTV, flags);
}

static inline void attn_launch(hipStream_t st, const float* qkv, float* o,
                               int VT, int w, int c)
{
  int blocks = (VT / w) * 6;
  int dh = c / 6;
  if (w == 8) {
    if (dh == 8)       k_attn<8, 8><<<blocks, 32, 0, st>>>(qkv, o, c);
    else if (dh == 16) k_attn<8, 16><<<blocks, 32, 0, st>>>(qkv, o, c);
    else               k_attn<8, 32><<<blocks, 32, 0, st>>>(qkv, o, c);
  } else {
    if (dh == 8)       k_attn<32, 8><<<blocks, 32, 0, st>>>(qkv, o, c);
    else if (dh == 16) k_attn<32, 16><<<blocks, 32, 0, st>>>(qkv, o, c);
    else               k_attn<32, 32><<<blocks, 32, 0, st>>>(qkv, o, c);
  }
}

extern "C" void kernel_launch(void* const* d_in, const int* in_sizes, int n_in,
                              void* d_out, int out_size, void* d_ws, size_t ws_size,
                              hipStream_t stream)
{
  (void)in_sizes; (void)n_in; (void)out_size; (void)ws_size;
  const float* x = (const float*)d_in[0];

  // ---- parameter walk (setup_inputs depth-first insertion order) ----
  int pi = 1;
  auto F = [&]() -> const float* { return (const float*)d_in[pi++]; };
  auto rdBN  = [&](BN& o)  { o.g = F(); o.b = F(); };
  auto rdGCN = [&](GCN& o) { o.w = F(); o.b = F(); rdBN(o.bn); };
  auto rdTCN = [&](TCN& o) {
    for (int i = 0; i < 4; ++i) {
      o.br[i].w1 = F(); o.br[i].b1 = F(); rdBN(o.br[i].bn1);
      o.br[i].wt = F(); o.br[i].bt = F(); rdBN(o.br[i].bn2);
    }
    o.mw1 = F(); o.mb1 = F(); rdBN(o.mbn1); rdBN(o.mbn2);
    o.ow = F(); o.ob = F(); rdBN(o.obn);
  };
  auto rdG3D = [&](G3D& o) {
    o.Ares = F(); o.w = F(); o.b = F(); rdBN(o.bn1);
    o.ow = F(); o.ob = F(); rdBN(o.bn2);
  };
  auto rdMOE = [&](MOE& o) {
    for (int e = 0; e < 2; ++e) {
      o.ex[e].lng = F(); o.ex[e].lnb = F();
      o.ex[e].wqkv = F(); o.ex[e].bqkv = F();
      o.ex[e].wo = F(); o.ex[e].bo = F();
    }
    o.gw = F(); o.gb = F(); o.pw = F(); o.pb = F(); rdBN(o.bn);
  };

  BN data_bn; rdBN(data_bn);
  GCN gcn1; rdGCN(gcn1);  TCN tcn_s1; rdTCN(tcn_s1);  MOE moe_s1; rdMOE(moe_s1);
  G3D g3d1a; rdG3D(g3d1a); G3D g3d1b; rdG3D(g3d1b);
  MOE tcn1m; rdMOE(tcn1m);
  GCN gcn2; rdGCN(gcn2);  MOE m1_2; rdMOE(m1_2);  MOE m2_2; rdMOE(m2_2);
  G3D g3d2a; rdG3D(g3d2a); G3D g3d2b; rdG3D(g3d2b);
  MOE tcn2m; rdMOE(tcn2m);
  GCN gcn3; rdGCN(gcn3);  MOE m1_3; rdMOE(m1_3);  MOE m2_3; rdMOE(m2_3);
  G3D g3d3a; rdG3D(g3d3a); G3D g3d3b; rdG3D(g3d3b);
  TCN tcn_s3; rdTCN(tcn_s3);
  const float* fcw = F(); const float* fcb = F();

  // ---- workspace bump allocator (floats) ----
  size_t off = 0;
  auto alloc = [&](size_t n) -> float* {
    float* p = (float*)d_ws + off;
    off += (n + 63) & ~(size_t)63;
    return p;
  };
  float* adjG  = alloc(13 * 17 * 17);
  float* adj3  = alloc(6 * 51 * 51);
  float* adj5  = alloc(6 * 85 * 85);
  float* aeff  = alloc(6 * 85 * 85);
  float* h0    = alloc(16 * 3 * 256 * VJ);
  float* bufS  = alloc(3400000);
  float* bufG  = alloc(3400000);
  float* bufH  = alloc(3400000);
  float* bufH2 = alloc(3400000);
  float* hb    = alloc(600000);
  float* aggb  = alloc(3200000);
  float* ubuf  = alloc(600000);
  float* hemb  = alloc(1100000);
  float* tok   = alloc(220000);
  float* tn    = alloc(220000);
  float* qkvb  = alloc(700000);
  float* ob    = alloc(220000);
  float* o2b   = alloc(220000);
  float* mixn  = alloc(220000);
  float* ybn   = alloc(450000);
  float* meanb = alloc(16 * 192);
  float* gateb = alloc(64);
  float* feat  = alloc(16 * 192);
  // f16 weight staging buffers (sizes in halves/2 floats, padded)
  _Float16* whA = (_Float16*)alloc(61000);   // up to 122000 halves (gcn/g3d/tcn w)
  _Float16* wh0 = (_Float16*)alloc(56000);   // wqkv expert 0
  _Float16* wh1 = (_Float16*)alloc(19000);   // wo   expert 0
  _Float16* wh2 = (_Float16*)alloc(56000);   // wqkv expert 1
  _Float16* wh3 = (_Float16*)alloc(19000);   // wo   expert 1
  _Float16* wh4 = (_Float16*)alloc(19000);   // pw

  // ---- adjacency constants (built on device, deterministic) ----
  k_build_adj<<<1, 256, 0, stream>>>(adjG, 1, 13);
  k_build_adj<<<1, 256, 0, stream>>>(adj3, 3, 6);
  k_build_adj<<<1, 256, 0, stream>>>(adj5, 5, 6);

  // ---- data_bn + layout ----
  {
    int tot = 16 * 3 * 256 * VJ;
    k_data_bn<<<cdiv(tot, 256), 256, 0, stream>>>(x, data_bn.g, data_bn.b, h0);
  }

  // ---- pipeline building blocks ----
  auto run_gcn = [&](const float* X, const GCN& p, int Ci, int Co, int T, float* Y) {
    int TV = T * VJ;
    cvt_w(stream, p.w, whA, Co * 13 * Ci);
    for (int n = 0; n < 16; ++n) {
      int tot = 13 * Ci * T * VJ;
      k_adj_agg<<<cdiv(tot, 256), 256, 0, stream>>>(
          adjG, X + (size_t)n * Ci * TV, aggb, Ci, T, VJ, 13);
      launch_gemm(stream, aggb, whA, p.b, p.bn.g, p.bn.b, nullptr,
                  Y + (size_t)n * Co * TV, TV, Co, 13 * Ci,
                  1, 13 * Ci, TV, 1, Co, 0, TV, 1 | 2 | 4);
    }
  };

  auto run_tcn = [&](const float* X, const TCN& p, int C, int T, float* Y) {
    int bc = C / 6, TV = T * VJ, rows = 16 * TV;
    int te = 16 * bc * T * VJ;
    for (int i = 0; i < 4; ++i) {
      const TBR& br = p.br[i];
      cvt_w(stream, br.w1, whA, bc * C);
      launch_gemm(stream, X, whA, br.b1, br.bn1.g, br.bn1.b, nullptr, hb,
                  rows, bc, C, 1, C, TV, 1, bc, 0, TV, 1 | 2 | 4);
      k_tconv<<<cdiv(te, 256), 256, 0, stream>>>(hb, br.wt, br.bt, br.bn2.g,
                                                 br.bn2.b, Y, bc, T, i + 1, C, i * bc);
    }
    cvt_w(stream, p.mw1, whA, bc * C);
    launch_gemm(stream, X, whA, p.mb1, p.mbn1.g, p.mbn1.b, nullptr, hb,
                rows, bc, C, 1, C, TV, 1, bc, 0, TV, 1 | 2 | 4);
    k_maxpool_bn<<<cdiv(te, 256), 256, 0, stream>>>(hb, p.mbn2.g, p.mbn2.b, Y,
                                                    bc, T, C, 4 * bc);
    cvt_w(stream, p.ow, whA, bc * C);
    launch_gemm(stream, X, whA, p.ob, p.obn.g, p.obn.b, nullptr, Y,
                rows, bc, C, 1, C, TV, 1, C, 5 * bc, TV, 1 | 2);
    int tot = 16 * C * TV;
    k_add_inplace<<<cdiv(tot, 256), 256, 0, stream>>>(Y, X, tot);
  };

  auto run_moe = [&](const float* X, const MOE& p, int Ci, int Co, int T,
                     int merge, int relu, float* Y) {
    int TV = T * VJ, VT = VJ * T;
    int Tout = merge ? T / 2 : T;
    int has_res = (!merge && Co == Ci) ? 1 : 0;
    const int wws[2] = {8, 32};
    _Float16* whq[2] = { wh0, wh2 };
    _Float16* who[2] = { wh1, wh3 };
    cvt_w(stream, p.ex[0].wqkv, wh0, 3 * Ci * Ci);
    cvt_w(stream, p.ex[0].wo,   wh1, Ci * Ci);
    cvt_w(stream, p.ex[1].wqkv, wh2, 3 * Ci * Ci);
    cvt_w(stream, p.ex[1].wo,   wh3, Ci * Ci);
    cvt_w(stream, p.pw,         wh4, Co * Ci);
    k_mean_tv<<<cdiv(16 * Ci, 256), 256, 0, stream>>>(X, meanb, Ci, TV);
    k_gate<<<1, 32, 0, stream>>>(meanb, p.gw, p.gb, gateb, Ci);
    for (int n = 0; n < 16; ++n) {
      const float* Xn = X + (size_t)n * Ci * TV;
      k_nctv_to_tok<<<cdiv(VT * Ci, 256), 256, 0, stream>>>(Xn, tok, Ci, T);
      for (int e = 0; e < 2; ++e) {
        const EXP& ex = p.ex[e];
        k_ln<<<VT, 32, 0, stream>>>(tok, ex.lng, ex.lnb, tn, Ci);
        launch_gemm(stream, tn, whq[e], ex.bqkv, nullptr, nullptr, nullptr,
                    qkvb, VT, 3 * Ci, Ci, 0, 0, 0, 0, 3 * Ci, 0, 0, 1);
        attn_launch(stream, qkvb, ob, VT, wws[e], Ci);
        launch_gemm(stream, ob, who[e], ex.bo, nullptr, nullptr, tok,
                    o2b, VT, Ci, Ci, 0, 0, 0, 0, Ci, 0, 0, 1 | 8);
        k_tok_accum<<<cdiv(VT * Ci, 256), 256, 0, stream>>>(
            o2b, gateb, n * 2 + e, mixn, VT * Ci, e == 0);
      }
      launch_gemm(stream, mixn, wh4, p.pb, nullptr, nullptr, nullptr,
                  ybn, VT, Co, Ci, 0, 0, 0, 0, Co, 0, 0, 1);
      int tot = Co * Tout * VJ;
      k_moe_post<<<cdiv(tot, 256), 256, 0, stream>>>(
          ybn, p.bn.g, p.bn.b, has_res ? Xn : nullptr,
          Y + (size_t)n * Co * Tout * VJ, Co, T, merge, relu, has_res);
    }
  };

  auto run_g3d = [&](const float* X, const G3D& p, int Ci, int Co, int wsz,
                     int stride, int T, int acc, float* gsum) {
    int Wd = wsz * VJ, pad = (wsz - 1) / 2;
    int tout = (T + 2 * pad - wsz) / stride + 1;
    const float* Abase = (wsz == 3) ? adj3 : adj5;
    int asz = 6 * Wd * Wd;
    k_axpy<<<cdiv(asz, 256), 256, 0, stream>>>(aeff, Abase, p.Ares, asz);
    int eo = (Ci == 3) ? Co : Ci;
    cvt_w(stream, p.w, whA, eo * 6 * Ci);
    for (int n = 0; n < 16; ++n) {
      const float* Xn = X + (size_t)n * Ci * T * VJ;
      int ut = Ci * tout * Wd;
      k_unfold<<<cdiv(ut, 256), 256, 0, stream>>>(Xn, ubuf, Ci, T, tout, wsz, stride, pad);
      int at = 6 * Ci * tout * Wd;
      k_adj_agg<<<cdiv(at, 256), 256, 0, stream>>>(aeff, ubuf, aggb, Ci, tout, Wd, 6);
      launch_gemm(stream, aggb, whA, p.b, p.bn1.g, p.bn1.b, nullptr, hemb,
                  tout * Wd, eo, 6 * Ci, 1, 6 * Ci, tout * Wd,
                  1, eo, 0, tout * Wd, 1 | 2 | 4);
      int ot = Co * tout * VJ;
      k_g3d_out<<<cdiv(ot, 256), 256, 0, stream>>>(
          hemb, p.ow, p.ob, p.bn2.g, p.bn2.b,
          gsum + (size_t)n * Co * tout * VJ, eo, Co, tout, wsz, acc);
    }
  };

  // ---- stage 1 (C=48, T=256) ----
  run_gcn(h0, gcn1, 3, 48, 256, bufS);
  run_tcn(bufS, tcn_s1, 48, 256, bufH2);
  run_moe(bufH2, moe_s1, 48, 48, 256, 0, 0, bufS);
  run_g3d(h0, g3d1a, 3, 48, 3, 1, 256, 0, bufG);
  run_g3d(h0, g3d1b, 3, 48, 5, 1, 256, 1, bufG);
  {
    int tot = 16 * 48 * 256 * VJ;
    k_add_relu<<<cdiv(tot, 256), 256, 0, stream>>>(bufS, bufG, bufH, tot);
  }
  run_moe(bufH, tcn1m, 48, 48, 256, 0, 1, bufH2);

  // ---- stage 2 (48 -> 96, T 256 -> 128) ----
  run_gcn(bufH2, gcn2, 48, 48, 256, bufS);
  run_moe(bufS, m1_2, 48, 96, 256, 1, 1, bufH);
  run_moe(bufH, m2_2, 96, 96, 128, 0, 0, bufS);
  run_g3d(bufH2, g3d2a, 48, 96, 3, 2, 256, 0, bufG);
  run_g3d(bufH2, g3d2b, 48, 96, 5, 2, 256, 1, bufG);
  {
    int tot = 16 * 96 * 128 * VJ;
    k_add_relu<<<cdiv(tot, 256), 256, 0, stream>>>(bufS, bufG, bufH, tot);
  }
  run_moe(bufH, tcn2m, 96, 96, 128, 0, 1, bufH2);

  // ---- stage 3 (96 -> 192, T 128 -> 64) ----
  run_gcn(bufH2, gcn3, 96, 96, 128, bufS);
  run_moe(bufS, m1_3, 96, 192, 128, 1, 1, bufH);
  run_moe(bufH, m2_3, 192, 192, 64, 0, 0, bufS);
  run_g3d(bufH2, g3d3a, 96, 192, 3, 2, 128, 0, bufG);
  run_g3d(bufH2, g3d3b, 96, 192, 5, 2, 128, 1, bufG);
  {
    int tot = 16 * 192 * 64 * VJ;
    k_add_relu<<<cdiv(tot, 256), 256, 0, stream>>>(bufS, bufG, bufH, tot);
  }
  run_tcn(bufH, tcn_s3, 192, 64, bufH2);

  // ---- head ----
  k_pool_feat<<<cdiv(8 * 192, 256), 256, 0, stream>>>(bufH2, feat, 192, 64);
  k_fc<<<cdiv(8 * 60, 32), 32, 0, stream>>>(feat, fcw, fcb, (float*)d_out, 192);
}
